// CausalAttention_49873160241896
// MI455X (gfx1250) — matrix operand order repaired and checked by reading, not compiled, not run on hardware
//
#include <hip/hip_runtime.h>

typedef __attribute__((ext_vector_type(16))) __bf16 v16bf;
typedef __attribute__((ext_vector_type(8)))  __bf16 v8bf;
typedef __attribute__((ext_vector_type(8)))  float  v8f;
typedef __attribute__((ext_vector_type(4)))  float  v4f;

namespace {
constexpr int kBatch = 8;
constexpr int kSeq   = 4096;
constexpr int kDin   = 768;
constexpr int kDout  = 64;
// scores * 64^-0.05 * log2(e): fold score scale and exp->exp2 conversion
constexpr float kC2 = 1.1718329f;

__device__ __forceinline__ v8f wmma_bf16(v16bf a, v16bf b, v8f c) {
  // D = A(16x32 bf16) x B(32x16 bf16) + C(16x16 f32)
  return __builtin_amdgcn_wmma_f32_16x16x32_bf16(false, a, false, b, (short)0, c,
                                                 false, false);
}

// A-fragment (16x32, 16-bit): lane holds K = e + (e>=8?8:0) (+8 if hi half,
// already folded into p). Two contiguous 16B chunks at p[0..7] and p[16..23].
__device__ __forceinline__ v16bf load_a16(const __bf16* p) {
  v8bf lo = *(const v8bf*)(p);
  v8bf hi = *(const v8bf*)(p + 16);
  v16bf r;
#pragma unroll
  for (int i = 0; i < 8; ++i) { r[i] = lo[i]; r[i + 8] = hi[i]; }
  return r;
}

// B-fragment (32x16, 16-bit): lane holds K = e (+16 if hi half, folded into p).
// 16 contiguous bf16.
__device__ __forceinline__ v16bf load_b16(const __bf16* p) {
  v8bf lo = *(const v8bf*)(p);
  v8bf hi = *(const v8bf*)(p + 8);
  v16bf r;
#pragma unroll
  for (int i = 0; i < 8; ++i) { r[i] = lo[i]; r[i + 8] = hi[i]; }
  return r;
}
} // namespace

// ---- Kernel 1: W[768][64] f32 -> Wt[64][768] bf16 (x3) --------------------
__global__ __launch_bounds__(256) void wconv_kernel(
    const float* __restrict__ Wq, const float* __restrict__ Wk,
    const float* __restrict__ Wv, __bf16* __restrict__ WtQ,
    __bf16* __restrict__ WtK, __bf16* __restrict__ WtV) {
  const int per = kDin * kDout;
  const int id = blockIdx.x * 256 + threadIdx.x;
  if (id >= 3 * per) return;
  const int which = id / per;
  const int r = id - which * per;
  const int n = r / kDin;
  const int k = r - n * kDin;
  const float* src = (which == 0) ? Wq : (which == 1 ? Wk : Wv);
  __bf16* dst = (which == 0) ? WtQ : (which == 1 ? WtK : WtV);
  dst[r] = (__bf16)src[k * kDout + n];
}

// ---- Kernel 2: Q/K/V projections, one wave per 16 rows --------------------
// Outputs: Qb,Kb row-major [B*S][64] bf16; Vt transposed [B][64][S] bf16.
__global__ __launch_bounds__(256) void proj_kernel(
    const float* __restrict__ x, const __bf16* __restrict__ WtQ,
    const __bf16* __restrict__ WtK, const __bf16* __restrict__ WtV,
    __bf16* __restrict__ Qb, __bf16* __restrict__ Kb,
    __bf16* __restrict__ Vt) {
  const int lane = threadIdx.x & 31;
  const int wave = threadIdx.x >> 5;
  const int tile = blockIdx.x * 8 + wave;   // 0..2047 row tiles of 16
  const int row0 = tile * 16;
  const int lrow = lane & 15;
  const bool hih = lane >= 16;
  const int achi = hih ? 16 : 0;            // A-fragment k-offset for hi half
  const int bchi = hih ? 16 : 0;            // B-fragment k-offset for hi half

  const v8f vzero = {};
  v8f accQ[4], accK[4], accV[4];
#pragma unroll
  for (int t = 0; t < 4; ++t) { accQ[t] = vzero; accK[t] = vzero; accV[t] = vzero; }

  const float* xrow = x + (row0 + lrow) * kDin + achi;
  const __bf16* wq = WtQ + lrow * kDin + bchi;
  const __bf16* wk = WtK + lrow * kDin + bchi;
  const __bf16* wv = WtV + lrow * kDin + bchi;

  for (int kb = 0; kb < kDin; kb += 32) {
    const float* px = xrow + kb;
    v4f x0 = *(const v4f*)(px);
    v4f x1 = *(const v4f*)(px + 4);
    v4f x2 = *(const v4f*)(px + 8);
    v4f x3 = *(const v4f*)(px + 12);
    v16bf a;
#pragma unroll
    for (int i = 0; i < 4; ++i) {
      a[i]      = (__bf16)x0[i];
      a[i + 4]  = (__bf16)x1[i];
      a[i + 8]  = (__bf16)x2[i];
      a[i + 12] = (__bf16)x3[i];
    }
#pragma unroll
    for (int nt = 0; nt < 4; ++nt) {
      const int off = nt * 16 * kDin + kb;
      accQ[nt] = wmma_bf16(a, load_b16(wq + off), accQ[nt]);
      accK[nt] = wmma_bf16(a, load_b16(wk + off), accK[nt]);
      accV[nt] = wmma_bf16(a, load_b16(wv + off), accV[nt]);
    }
  }

  // C-layout: lane holds column d = nt*16+lrow, rows row0 + soff + r
  const int soff = hih ? 8 : 0;
  const int bb = row0 >> 12;                // batch (4096 rows each)
  const int sl = (row0 & 4095) + soff;      // seq pos within batch
#pragma unroll
  for (int nt = 0; nt < 4; ++nt) {
    const int d = nt * 16 + lrow;
#pragma unroll
    for (int r = 0; r < 8; ++r) {
      const int s = row0 + soff + r;
      Qb[s * kDout + d] = (__bf16)accQ[nt][r];
      Kb[s * kDout + d] = (__bf16)accK[nt][r];
    }
    v8bf pv;
#pragma unroll
    for (int r = 0; r < 8; ++r) pv[r] = (__bf16)accV[nt][r];
    *(v8bf*)(Vt + bb * (kDout * kSeq) + d * kSeq + sl) = pv;  // 16B contiguous
  }
}

// ---- Kernel 3: flash attention, one wave per 16 queries -------------------
__global__ __launch_bounds__(256) void attn_kernel(
    const __bf16* __restrict__ Qb, const __bf16* __restrict__ Kb,
    const __bf16* __restrict__ Vt, float* __restrict__ out) {
  const int lane = threadIdx.x & 31;
  const int wave = threadIdx.x >> 5;
  const int tile = blockIdx.x * 8 + wave;   // 0..2047 query tiles
  const int b    = tile >> 8;               // 256 tiles per batch
  const int q0   = (tile & 255) * 16;
  const int lrow = lane & 15;
  const bool hih = lane >= 16;

  // Q^T B-fragments (d-slices 0:32 and 32:64), loaded once per wave
  const __bf16* qp = Qb + ((b * kSeq) + q0 + lrow) * kDout + (hih ? 8 : 0);
  const v16bf bq0 = load_a16(qp);
  const v16bf bq1 = load_a16(qp + 32);

  const __bf16* kp  = Kb + ((b * kSeq) + lrow) * kDout + (hih ? 8 : 0);
  const __bf16* vpb = Vt + b * (kDout * kSeq) + lrow * kSeq + (hih ? 8 : 0);
  const __bf16* vp0 = vpb;
  const __bf16* vp1 = vpb + 16 * kSeq;
  const __bf16* vp2 = vpb + 32 * kSeq;
  const __bf16* vp3 = vpb + 48 * kSeq;

  const v8f vzero = {};
  v8f O[4];
#pragma unroll
  for (int t = 0; t < 4; ++t) O[t] = vzero;
  float m_run = -1e30f, l_run = 0.0f;

  for (int kb = 0; kb < kSeq; kb += 32) {
    // S^T tiles: [key 0..15 | 16..31] x [16 queries], reduce over d=64
    const __bf16* krow = kp + kb * kDout;
    v8f s0 = vzero, s1 = vzero;
    s0 = wmma_bf16(load_a16(krow), bq0, s0);
    s0 = wmma_bf16(load_a16(krow + 32), bq1, s0);
    s1 = wmma_bf16(load_a16(krow + 16 * kDout), bq0, s1);
    s1 = wmma_bf16(load_a16(krow + 16 * kDout + 32), bq1, s1);

    // per-query online softmax; query q lives in lanes {q, q+16}
    float mloc = -1e30f;
    v8f t0, t1;
#pragma unroll
    for (int r = 0; r < 8; ++r) {
      t0[r] = s0[r] * kC2;
      t1[r] = s1[r] * kC2;
      mloc = fmaxf(mloc, fmaxf(t0[r], t1[r]));
    }
    const float m_tile = fmaxf(mloc, __shfl_xor(mloc, 16, 32));
    const float m_new  = fmaxf(m_run, m_tile);
    const float alpha  = exp2f(m_run - m_new);

    v8f p0, p1;
    float lsum = 0.0f;
#pragma unroll
    for (int r = 0; r < 8; ++r) {
      p0[r] = exp2f(t0[r] - m_new);
      p1[r] = exp2f(t1[r] - m_new);
      lsum += p0[r] + p1[r];
    }
    lsum += __shfl_xor(lsum, 16, 32);
    l_run = l_run * alpha + lsum;
    m_run = m_new;

    // Convert exp(S^T) C-layout -> P^T B-fragment (32 keys x 16 q):
    // element e: key = e + (e>=8?8:0) + (hi?8:0); all values lane-local.
    v16bf pb;
#pragma unroll
    for (int r = 0; r < 8; ++r) {
      pb[r]     = (__bf16)p0[r];
      pb[r + 8] = (__bf16)p1[r];
    }

#pragma unroll
    for (int t = 0; t < 4; ++t) {
#pragma unroll
      for (int r = 0; r < 8; ++r) O[t][r] *= alpha;
    }
    // out^T[d][q] += Vt(16d x 32keys) x P^T(32keys x 16q)
    O[0] = wmma_bf16(load_a16(vp0 + kb), pb, O[0]);
    O[1] = wmma_bf16(load_a16(vp1 + kb), pb, O[1]);
    O[2] = wmma_bf16(load_a16(vp2 + kb), pb, O[2]);
    O[3] = wmma_bf16(load_a16(vp3 + kb), pb, O[3]);
  }

  // C-layout of out^T: lane holds q = q0+lrow, d = t*16 + (hi?8:0) + r
  const float inv_l = 1.0f / l_run;
  float* op = out + ((b * kSeq) + q0 + lrow) * kDout + (hih ? 8 : 0);
#pragma unroll
  for (int t = 0; t < 4; ++t) {
    v4f o0, o1;
#pragma unroll
    for (int i = 0; i < 4; ++i) {
      o0[i] = O[t][i] * inv_l;
      o1[i] = O[t][i + 4] * inv_l;
    }
    *(v4f*)(op + t * 16)     = o0;
    *(v4f*)(op + t * 16 + 4) = o1;
  }
}

extern "C" void kernel_launch(void* const* d_in, const int* in_sizes, int n_in,
                              void* d_out, int out_size, void* d_ws,
                              size_t ws_size, hipStream_t stream) {
  (void)in_sizes; (void)n_in; (void)out_size; (void)ws_size;
  const float* x  = (const float*)d_in[0];
  const float* Wq = (const float*)d_in[1];
  const float* Wk = (const float*)d_in[2];
  const float* Wv = (const float*)d_in[3];
  float* out = (float*)d_out;

  char* ws = (char*)d_ws;
  const size_t wBytes   = (size_t)kDin * kDout * sizeof(__bf16);            // 96 KB
  const size_t qkvBytes = (size_t)kBatch * kSeq * kDout * sizeof(__bf16);   // 4 MB
  __bf16* WtQ = (__bf16*)(ws);
  __bf16* WtK = (__bf16*)(ws + wBytes);
  __bf16* WtV = (__bf16*)(ws + 2 * wBytes);
  __bf16* Qb  = (__bf16*)(ws + 3 * wBytes);
  __bf16* Kb  = (__bf16*)(ws + 3 * wBytes + qkvBytes);
  __bf16* Vt  = (__bf16*)(ws + 3 * wBytes + 2 * qkvBytes);

  wconv_kernel<<<(3 * kDin * kDout + 255) / 256, 256, 0, stream>>>(
      Wq, Wk, Wv, WtQ, WtK, WtV);
  // 2048 row-tiles of 16, 8 waves per block
  proj_kernel<<<256, 256, 0, stream>>>(x, WtQ, WtK, WtV, Qb, Kb, Vt);
  // 2048 query-tiles of 16, 8 waves per block
  attn_kernel<<<256, 256, 0, stream>>>(Qb, Kb, Vt, out);
}
